// BrainModel_1640677507517
// MI455X (gfx1250) — matrix operand order repaired and checked by reading, not compiled, not run on hardware
//
#include <hip/hip_runtime.h>
#include <math.h>

typedef __attribute__((ext_vector_type(16))) _Float16 v16h;
typedef __attribute__((ext_vector_type(8)))  _Float16 v8h;
typedef __attribute__((ext_vector_type(2)))  __fp16   v2hf;   // cvt_pkrtz result type
typedef __attribute__((ext_vector_type(8)))  float    v8f;
typedef __attribute__((ext_vector_type(4)))  float    v4f;

#define WAVES   8
#define THREADS (WAVES * 32)

struct __attribute__((aligned(16))) WaveScratch {
    _Float16 xT[16][32];     // x tile, f16: cols 0-9 = x, col 10 = 1.0 (bias), 11-31 = 0
    _Float16 cT[16][32];     // concat tile: cols 0-1 fused, 2-11 x, col 12 = 1.0, rest 0
    _Float16 actT[16][128];  // gate/refine h (cols 0-63), expert h: 4x32-col ping-pong slots
    float    logits[16][8];
    float    outs[16][16];   // expert outputs: [token][e*2 + a]
    float    outStage[32];   // coalesced output staging (16 tokens x 2)
};

struct __attribute__((aligned(16))) Smem {
    // weights transposed [out][in], K padded to 32/64, f16; biases folded into
    // spare K rows where the layer has padding (L1 k=10, gate1 k=10, refine1 k=12)
    _Float16 W1t[8][32][32];   // expert L1: 10->32 (+bias row at k=10)
    _Float16 W2t[8][32][32];   // expert L2: 32->32
    _Float16 W3t[8][16][32];   // expert L3: 32->2 (out padded to 16)
    _Float16 G1t[64][32];      // gate L1: 10->64 (+bias row at k=10)
    _Float16 G2t[16][64];      // gate L2: 64->8 (out padded to 16)
    _Float16 R1t[64][32];      // refine L1: 12->64 (+bias row at k=12)
    _Float16 R2t[16][64];      // refine L2: 64->2 (out padded to 16)
    float b2[8][32];
    float b3[8][2];
    float gb2[8];
    float rb2[2];
    WaveScratch w[WAVES];
};

__device__ inline v8f zero8() {
    v8f z;
#pragma unroll
    for (int i = 0; i < 8; ++i) z[i] = 0.0f;
    return z;
}

// pack two f32 to f16x2 (v_cvt_pk_rtz_f16_f32), ReLU in packed f16 (v_pk_max),
// then scatter the two halves (b16 / b16_d16_hi stores).
__device__ inline void pk_relu2(float a, float b, _Float16& lo, _Float16& hi) {
    v2hf pk = __builtin_amdgcn_cvt_pkrtz(a, b);
    v2hf z  = { (__fp16)0.0f, (__fp16)0.0f };
    v2hf r  = __builtin_elementwise_max(pk, z);
    union { v2hf v; _Float16 h[2]; } u;
    u.v = r;
    lo = u.h[0];
    hi = u.h[1];
}

// A fragment (16x32 K-slice), 16-bit layout per ISA 7.12.2:
// lanes 0-15: halves 0-7 = K0..7, halves 8-15 = K16..23 ; lanes 16-31: K8..15 / K24..31
__device__ inline v16h load_A(const _Float16* tile, int ldRow, int kBase, int lane) {
    int m  = lane & 15;
    int k0 = kBase + ((lane < 16) ? 0 : 8);
    const _Float16* row = tile + m * ldRow;
    union { v16h v; v8h h[2]; } u;
    u.h[0] = *(const v8h*)(row + k0);
    u.h[1] = *(const v8h*)(row + k0 + 16);
    return u.v;
}

// B fragment (32x16 slice) from transposed weights Wt[n][k]:
// lane n = nBase + lane%16; lanes 0-15 hold K kBase+0..7,+16..23, lanes 16-31 kBase+8..15,+24..31
__device__ inline v16h load_B(const _Float16* wt, int ldRow, int nBase, int kBase, int lane) {
    int n  = nBase + (lane & 15);
    int k0 = kBase + ((lane < 16) ? 0 : 8);
    const _Float16* row = wt + n * ldRow;
    union { v16h v; v8h h[2]; } u;
    u.h[0] = *(const v8h*)(row + k0);
    u.h[1] = *(const v8h*)(row + k0 + 16);
    return u.v;
}

__device__ inline v8f wmma_f16(v16h a, v16h b, v8f c) {
    return __builtin_amdgcn_wmma_f32_16x16x32_f16(false, a, false, b, (short)0, c, false, false);
}

// C/D layout: lane L element j -> row m = j + (L<16?0:8), col n = nBase + (L&15)
__device__ inline void store_act_relu(v8f c, _Float16* tile, int ldRow, int nBase, int lane) {
    int n     = nBase + (lane & 15);
    int mBase = (lane < 16) ? 0 : 8;
#pragma unroll
    for (int j = 0; j < 8; j += 2) {
        _Float16 lo, hi;
        pk_relu2(c[j], c[j + 1], lo, hi);
        tile[(mBase + j + 0) * ldRow + n] = lo;
        tile[(mBase + j + 1) * ldRow + n] = hi;
    }
}

__device__ inline void store_act_relu_bias(v8f c, _Float16* tile, int ldRow, int nBase,
                                           const float* bias, int lane) {
    int n     = nBase + (lane & 15);
    int mBase = (lane < 16) ? 0 : 8;
    float bv  = bias[n];
#pragma unroll
    for (int j = 0; j < 8; j += 2) {
        _Float16 lo, hi;
        pk_relu2(c[j] + bv, c[j + 1] + bv, lo, hi);
        tile[(mBase + j + 0) * ldRow + n] = lo;
        tile[(mBase + j + 1) * ldRow + n] = hi;
    }
}

__global__ void __launch_bounds__(THREADS)
moe_fused_kernel(const float* __restrict__ x,
                 const float* __restrict__ W1, const float* __restrict__ b1,
                 const float* __restrict__ W2, const float* __restrict__ b2,
                 const float* __restrict__ W3, const float* __restrict__ b3,
                 const float* __restrict__ G1, const float* __restrict__ gb1,
                 const float* __restrict__ G2, const float* __restrict__ gb2,
                 const float* __restrict__ R1, const float* __restrict__ rb1,
                 const float* __restrict__ R2, const float* __restrict__ rb2,
                 float* __restrict__ out, int nTokens) {
    extern __shared__ char smem_raw[];
    Smem& S = *reinterpret_cast<Smem*>(smem_raw);

    const int tid = threadIdx.x;

    // ---- stage weights (transposed, f16, zero-padded, biases in spare K rows) ----
    for (int i = tid; i < 8 * 32 * 32; i += THREADS) {   // W1t[e][n][k]
        int e = i >> 10, r = i & 1023, n = r >> 5, k = r & 31;
        float v = (k < 10) ? W1[(e * 10 + k) * 32 + n]
                           : ((k == 10) ? b1[e * 32 + n] : 0.0f);
        (&S.W1t[0][0][0])[i] = (_Float16)v;
    }
    for (int i = tid; i < 8 * 32 * 32; i += THREADS) {   // W2t
        int e = i >> 10, r = i & 1023, n = r >> 5, k = r & 31;
        (&S.W2t[0][0][0])[i] = (_Float16)W2[(e * 32 + k) * 32 + n];
    }
    for (int i = tid; i < 8 * 16 * 32; i += THREADS) {   // W3t (n padded 2->16)
        int e = i >> 9, r = i & 511, n = r >> 5, k = r & 31;
        float v = (n < 2) ? W3[(e * 32 + k) * 2 + n] : 0.0f;
        (&S.W3t[0][0][0])[i] = (_Float16)v;
    }
    for (int i = tid; i < 64 * 32; i += THREADS) {       // G1t
        int n = i >> 5, k = i & 31;
        float v = (k < 10) ? G1[k * 64 + n] : ((k == 10) ? gb1[n] : 0.0f);
        (&S.G1t[0][0])[i] = (_Float16)v;
    }
    for (int i = tid; i < 16 * 64; i += THREADS) {       // G2t (n padded 8->16)
        int n = i >> 6, k = i & 63;
        float v = (n < 8) ? G2[k * 8 + n] : 0.0f;
        (&S.G2t[0][0])[i] = (_Float16)v;
    }
    for (int i = tid; i < 64 * 32; i += THREADS) {       // R1t (K real 12, bias at k=12)
        int n = i >> 5, k = i & 31;
        float v = (k < 12) ? R1[k * 64 + n] : ((k == 12) ? rb1[n] : 0.0f);
        (&S.R1t[0][0])[i] = (_Float16)v;
    }
    for (int i = tid; i < 16 * 64; i += THREADS) {       // R2t (n padded 2->16)
        int n = i >> 6, k = i & 63;
        float v = (n < 2) ? R2[k * 2 + n] : 0.0f;
        (&S.R2t[0][0])[i] = (_Float16)v;
    }
    for (int i = tid; i < 256; i += THREADS) (&S.b2[0][0])[i] = b2[i];
    for (int i = tid; i < 16;  i += THREADS) (&S.b3[0][0])[i] = b3[i];
    if (tid < 8) S.gb2[tid] = gb2[tid];
    if (tid < 2) S.rb2[tid] = rb2[tid];
    __syncthreads();

    const int wave = tid >> 5;
    const int lane = tid & 31;
    WaveScratch& W = S.w[wave];

    const int mBase = (lane < 16) ? 0 : 8;
    const int nCol  = lane & 15;
    const int tiles = nTokens >> 4;

    // one-time per-wave init: x-tile pads (col 10 = 1.0 feeds bias rows),
    // concat-tile pads (col 12 = 1.0 feeds refine bias row)
    for (int i = lane; i < 16 * 32; i += 32) {
        int d = i & 31;
        (&W.xT[0][0])[i] = (d == 10) ? (_Float16)1.0f : (_Float16)0.0f;
    }
    if (lane < 16) {
#pragma unroll
        for (int d = 12; d < 32; ++d)
            W.cT[lane][d] = (d == 12) ? (_Float16)1.0f : (_Float16)0.0f;
    }

    for (int tile = blockIdx.x * WAVES + wave; tile < tiles; tile += gridDim.x * WAVES) {
        const int base = tile << 4;

        // ---- stage x tile: 160 contiguous floats = 40 coalesced float4 loads ----
        {
            const v4f* xv = (const v4f*)(x + (size_t)base * 10);  // 640B tile: 16B aligned
#pragma unroll
            for (int r = 0; r < 2; ++r) {
                int i = lane + r * 32;
                if (i < 40) {
                    v4f v = xv[i];
#pragma unroll
                    for (int q = 0; q < 4; ++q) {
                        int idx = i * 4 + q;
                        W.xT[idx / 10][idx % 10] = (_Float16)v[q];
                    }
                }
            }
        }

        v16h Ax = load_A(&W.xT[0][0], 32, 0, lane);

        // ---- gating L1: 10->64 (bias folded into k=10 row) ----
#pragma unroll
        for (int nt = 0; nt < 4; ++nt) {
            v16h Bg = load_B(&S.G1t[0][0], 32, nt * 16, 0, lane);
            v8f  cg = wmma_f16(Ax, Bg, zero8());
            store_act_relu(cg, &W.actT[0][0], 128, nt * 16, lane);
        }
        // ---- gating L2: 64->8, K split into two chained WMMAs ----
        {
            v16h Ag0 = load_A(&W.actT[0][0], 128, 0, lane);
            v16h Ag1 = load_A(&W.actT[0][0], 128, 32, lane);
            v8f  cl  = wmma_f16(Ag0, load_B(&S.G2t[0][0], 64, 0, 0, lane), zero8());
            cl       = wmma_f16(Ag1, load_B(&S.G2t[0][0], 64, 0, 32, lane), cl);
            if (nCol < 8) {
                float bv = S.gb2[nCol];
#pragma unroll
                for (int j = 0; j < 8; ++j) W.logits[mBase + j][nCol] = cl[j] + bv;
            }
        }

        // ---- experts: 4 independent 32-col h slots so 4 consecutive experts
        //      have independent wmma->lds->wmma chains ----
#pragma unroll 4
        for (int e = 0; e < 8; ++e) {
            const int off = (e & 3) * 32;
            // L1: 10->32 (bias folded into k=10 row)
            v8f c0 = wmma_f16(Ax, load_B(&S.W1t[e][0][0], 32, 0, 0, lane), zero8());
            v8f c1 = wmma_f16(Ax, load_B(&S.W1t[e][0][0], 32, 16, 0, lane), zero8());
            store_act_relu(c0, &W.actT[0][0], 128, off + 0, lane);
            store_act_relu(c1, &W.actT[0][0], 128, off + 16, lane);
            // L2: 32->32
            v16h Ah = load_A(&W.actT[0][0], 128, off, lane);
            c0 = wmma_f16(Ah, load_B(&S.W2t[e][0][0], 32, 0, 0, lane), zero8());
            c1 = wmma_f16(Ah, load_B(&S.W2t[e][0][0], 32, 16, 0, lane), zero8());
            store_act_relu_bias(c0, &W.actT[0][0], 128, off + 0, S.b2[e], lane);
            store_act_relu_bias(c1, &W.actT[0][0], 128, off + 16, S.b2[e], lane);
            // L3: 32->2 (N padded to 16), tanh
            v16h Ah2 = load_A(&W.actT[0][0], 128, off, lane);
            v8f  c3  = wmma_f16(Ah2, load_B(&S.W3t[e][0][0], 32, 0, 0, lane), zero8());
            if (nCol < 2) {
                float bv = S.b3[e][nCol];
#pragma unroll
                for (int j = 0; j < 8; ++j)
                    W.outs[mBase + j][e * 2 + nCol] = tanhf(c3[j] + bv);
            }
        }

        // ---- softmax over experts + weighted fusion + concat (lanes 0-15) ----
        if (lane < 16) {
            int m = lane;
            float l[8], mx = -1e30f;
#pragma unroll
            for (int e = 0; e < 8; ++e) { l[e] = W.logits[m][e]; mx = fmaxf(mx, l[e]); }
            float s = 0.0f;
#pragma unroll
            for (int e = 0; e < 8; ++e) { l[e] = __expf(l[e] - mx); s += l[e]; }
            float inv = 1.0f / s, f0 = 0.0f, f1 = 0.0f;
#pragma unroll
            for (int e = 0; e < 8; ++e) {
                float we = l[e] * inv;
                f0 += W.outs[m][e * 2 + 0] * we;
                f1 += W.outs[m][e * 2 + 1] * we;
            }
            W.cT[m][0] = (_Float16)f0;
            W.cT[m][1] = (_Float16)f1;
#pragma unroll
            for (int d = 0; d < 10; ++d) W.cT[m][2 + d] = W.xT[m][d];
        }

        // ---- refine L1: 12->64 (bias folded into k=12 row) ----
        v16h Ac = load_A(&W.cT[0][0], 32, 0, lane);
#pragma unroll
        for (int nt = 0; nt < 4; ++nt) {
            v16h Br = load_B(&S.R1t[0][0], 32, nt * 16, 0, lane);
            v8f  cr = wmma_f16(Ac, Br, zero8());
            store_act_relu(cr, &W.actT[0][0], 128, nt * 16, lane);
        }
        // ---- refine L2: 64->2, tanh, coalesced store ----
        {
            v16h Ar0 = load_A(&W.actT[0][0], 128, 0, lane);
            v16h Ar1 = load_A(&W.actT[0][0], 128, 32, lane);
            v8f  co  = wmma_f16(Ar0, load_B(&S.R2t[0][0], 64, 0, 0, lane), zero8());
            co       = wmma_f16(Ar1, load_B(&S.R2t[0][0], 64, 0, 32, lane), co);
            if (nCol < 2) {
                float bv = S.rb2[nCol];
#pragma unroll
                for (int j = 0; j < 8; ++j)
                    W.outStage[(mBase + j) * 2 + nCol] = tanhf(co[j] + bv);
            }
            // 16 tokens x 2 outputs = 32 contiguous floats: one lane-wide store
            out[(size_t)base * 2 + lane] = W.outStage[lane];
        }
    }
}

extern "C" void kernel_launch(void* const* d_in, const int* in_sizes, int n_in,
                              void* d_out, int out_size, void* d_ws, size_t ws_size,
                              hipStream_t stream) {
    const float* x   = (const float*)d_in[0];
    const float* W1  = (const float*)d_in[1];
    const float* b1  = (const float*)d_in[2];
    const float* W2  = (const float*)d_in[3];
    const float* b2  = (const float*)d_in[4];
    const float* W3  = (const float*)d_in[5];
    const float* b3  = (const float*)d_in[6];
    const float* G1  = (const float*)d_in[7];
    const float* gb1 = (const float*)d_in[8];
    const float* G2  = (const float*)d_in[9];
    const float* gb2 = (const float*)d_in[10];
    const float* R1  = (const float*)d_in[11];
    const float* rb1 = (const float*)d_in[12];
    const float* R2  = (const float*)d_in[13];
    const float* rb2 = (const float*)d_in[14];
    float* out = (float*)d_out;

    int nTokens = in_sizes[0] / 10;
    int tiles   = nTokens >> 4;
    int blocks  = 2048;
    int maxb    = (tiles + WAVES - 1) / WAVES;
    if (blocks > maxb) blocks = maxb;
    if (blocks < 1) blocks = 1;

    moe_fused_kernel<<<blocks, THREADS, (int)sizeof(Smem), stream>>>(
        x, W1, b1, W2, b2, W3, b3, G1, gb1, G2, gb2, R1, rb1, R2, rb2, out, nTokens);
}
